// SpatialSelfAttention_52029233824427
// MI455X (gfx1250) — matrix operand order repaired and checked by reading, not compiled, not run on hardware
//
#include <hip/hip_runtime.h>

// ---------------------------------------------------------------------------
// SpatialSelfAttention for MI455X (gfx1250, wave32, WMMA)
//   Q = H Wq + bq ; K = H Wk + bk       (bf16 workspace, Q pre-scaled)
//   A = softmax(zero_diag(Q K^T)/sqrt(128)) ; out = A H + H
// Fused flash-attention, O^T = V^T P^T formulation so every WMMA operand is a
// contiguous per-lane load (global b128 / ds b128), no fragment transposes.
// Round 4: fit under 256 VGPRs (no s_set_vgpr_msb churn) -- shallow register
// pipelines (K depth-2, V depth-3) + zero-register global_prefetch_b8 one
// key-block ahead for both K and V streams; base-2 online softmax.
// ---------------------------------------------------------------------------

typedef __bf16 bf16_t;
typedef __attribute__((ext_vector_type(16))) __bf16 v16bf;
typedef __attribute__((ext_vector_type(8)))  float  v8f;
typedef unsigned int u32x4 __attribute__((ext_vector_type(4)));
typedef float f32x4 __attribute__((ext_vector_type(4)));

union B16Frag { v16bf v; u32x4 q[2]; };

#define NN 16384
#define DK 128
#define DV 256
// 1/sqrt(128) * log2(e): makes S a base-2 logit -> softmax via exp2 only
#define QSCALE (0.08838834764831845f * 1.4426950408889634f)

// ---------------------------------------------------------------------------
// Kernel 1: per-row projections  Qb = (H Wq + bq)*QSCALE, Kb = H Wk + bk
// ---------------------------------------------------------------------------
__global__ __launch_bounds__(128) void proj_kernel(
    const float* __restrict__ H,  const float* __restrict__ Wq,
    const float* __restrict__ bq, const float* __restrict__ Wk,
    const float* __restrict__ bk, bf16_t* __restrict__ Qb,
    bf16_t* __restrict__ Kb)
{
    __shared__ float hrow[DV];
    const int row = blockIdx.x;
    const int t   = threadIdx.x;
    hrow[t]       = H[(size_t)row * DV + t];
    hrow[t + 128] = H[(size_t)row * DV + t + 128];
    __syncthreads();

    float aq = bq[t];
    float ak = bk[t];
#pragma unroll 8
    for (int k = 0; k < DV; ++k) {
        const float h = hrow[k];
        aq += h * Wq[k * DK + t];
        ak += h * Wk[k * DK + t];
    }
    Qb[(size_t)row * DK + t] = (bf16_t)(aq * QSCALE);
    Kb[(size_t)row * DK + t] = (bf16_t)ak;
}

// ---------------------------------------------------------------------------
// Kernel 2: HT[c][r] = (bf16) H[r][c]   (64x64 tiles through LDS)
// ---------------------------------------------------------------------------
__global__ __launch_bounds__(256) void transpose_kernel(
    const float* __restrict__ H, bf16_t* __restrict__ HT)
{
    __shared__ bf16_t tile[64][65];
    const int r0 = blockIdx.x * 64;
    const int c0 = blockIdx.y * 64;
    const int tc = threadIdx.x & 63;
    const int tr = threadIdx.x >> 6;      // 0..3
#pragma unroll
    for (int i = 0; i < 16; ++i) {
        const int r = tr + i * 4;
        tile[r][tc] = (bf16_t)H[(size_t)(r0 + r) * DV + c0 + tc];
    }
    __syncthreads();
#pragma unroll
    for (int i = 0; i < 16; ++i) {
        const int c = tr + i * 4;
        HT[(size_t)(c0 + c) * NN + r0 + tc] = tile[tc][c];
    }
}

// ---------------------------------------------------------------------------
// Kernel 3: fused flash attention (base-2 online softmax).
// blockDim = 128 (4 independent waves); wave w owns 16 query rows.
// ---------------------------------------------------------------------------
#define KLOAD(n, jj)                                                          \
    do {                                                                      \
        const bf16_t* kp_ =                                                   \
            kbase0 + (size_t)((jj) + 16 * ((n) >> 2)) * DK + ((n)&3) * 32;    \
        kb[(n)&1].q[0] = *(const u32x4*)(kp_);                                \
        kb[(n)&1].q[1] = *(const u32x4*)(kp_ + 16);                           \
    } while (0)

#define VLOAD(n)                                                              \
    do {                                                                      \
        const bf16_t* vp_ = vbase + (size_t)((n)*16) * NN + j0;               \
        va[(n) % 3].q[0] = *(const u32x4*)(vp_);                              \
        va[(n) % 3].q[1] = *(const u32x4*)(vp_ + 16);                         \
    } while (0)

__global__ __launch_bounds__(128) void attn_kernel(
    const float*  __restrict__ H,   const bf16_t* __restrict__ Qb,
    const bf16_t* __restrict__ Kb,  const bf16_t* __restrict__ HT,
    float* __restrict__ out)
{
    __shared__ __attribute__((aligned(16))) bf16_t Plds[4][16 * 32]; // 1 KB/wave
    __shared__ float Slds[4][16];                                    // bcast buf

    const int lane = threadIdx.x & 31;
    const int wave = threadIdx.x >> 5;
    const int half = lane >> 4;        // 0: lanes 0-15, 1: lanes 16-31
    const int lrow = lane & 15;
    const int q0   = (blockIdx.x * 4 + wave) * 16;

    bf16_t* pw = &Plds[wave][0];
    float*  sb = &Slds[wave][0];

    // per-lane base pointers (loop-invariant)
    const bf16_t* kbase0 = Kb + (size_t)lrow * DK + half * 8;   // + j*DK + f*32
    const bf16_t* vbase  = HT + (size_t)lrow * NN + half * 8;   // + ft*16*NN + j0

    // ---- preload Q A-fragments: 16x32 bf16 each, K = 0..127 ----------------
    v16bf qa[4];
#pragma unroll
    for (int f = 0; f < 4; ++f) {
        B16Frag u;
        const bf16_t* p = Qb + (size_t)(q0 + lrow) * DK + f * 32 + half * 8;
        u.q[0] = *(const u32x4*)(p);
        u.q[1] = *(const u32x4*)(p + 16);
        qa[f] = u.v;
    }

    // ---- O^T accumulators: 16 feature tiles (16f x 16q each) ---------------
    v8f acc[16];
    const v8f vzero = {0.f, 0.f, 0.f, 0.f, 0.f, 0.f, 0.f, 0.f};
#pragma unroll
    for (int ft = 0; ft < 16; ++ft) acc[ft] = vzero;

    float m_r[8], l_r[8];   // l_r: per-lane partial sums (reduced at the end)
#pragma unroll
    for (int r = 0; r < 8; ++r) { m_r[r] = -3.0e38f; l_r[r] = 0.f; }

    // ======================= key-block loop =================================
    for (int j0 = 0; j0 < NN; j0 += 32) {
        // ---- S = Q K^T: 8 fragments, depth-2 rotating buffer ---------------
        v8f s[2] = {vzero, vzero};
        B16Frag kb[2];
        KLOAD(0, j0);
#pragma unroll
        for (int i = 0; i < 8; ++i) {
            if (i < 7) KLOAD(i + 1, j0);
            s[i >> 2] = __builtin_amdgcn_wmma_f32_16x16x32_bf16(
                false, qa[i & 3], false, kb[i & 1].v, (short)0, s[i >> 2],
                false, false);
        }

        // ---- zero diagonal (pre-scale per reference; 0 stays 0) ------------
        if (j0 <= q0 + 15 && j0 + 31 >= q0) {
            const int kg0 = j0 + lrow;
            const int kg1 = j0 + 16 + lrow;
#pragma unroll
            for (int r = 0; r < 8; ++r) {
                const int qg = q0 + r + 8 * half;
                if (kg0 == qg) s[0][r] = 0.f;
                if (kg1 == qg) s[1][r] = 0.f;
            }
        }

        // ---- online softmax in base 2 (rows of S live across 16 lanes) -----
#pragma unroll
        for (int r = 0; r < 8; ++r) {
            float mx = fmaxf(s[0][r], s[1][r]);
            mx = fmaxf(mx, __shfl_xor(mx, 1, 32));
            mx = fmaxf(mx, __shfl_xor(mx, 2, 32));
            mx = fmaxf(mx, __shfl_xor(mx, 4, 32));
            mx = fmaxf(mx, __shfl_xor(mx, 8, 32));
            const float mn = fmaxf(m_r[r], mx);
            const float al = __builtin_amdgcn_exp2f(m_r[r] - mn);
            m_r[r] = mn;
            const float p0 = __builtin_amdgcn_exp2f(s[0][r] - mn);
            const float p1 = __builtin_amdgcn_exp2f(s[1][r] - mn);
            s[0][r] = p0;
            s[1][r] = p1;
            l_r[r] = l_r[r] * al + (p0 + p1);   // per-lane partial row sum
            if (lrow == r) sb[r + 8 * half] = al;   // publish rescale factor
        }

        // ---- store P (bf16) row-major [16 q][32 k] into wave-private LDS ---
#pragma unroll
        for (int r = 0; r < 8; ++r) {
            const int qr = r + 8 * half;
            pw[qr * 32 + lrow]      = (bf16_t)s[0][r];
            pw[qr * 32 + 16 + lrow] = (bf16_t)s[1][r];
        }
        // LDS is in-order within a wave: loads issued below see the stores.
        const float alane = sb[lrow];              // rescale factor for q=lrow
        B16Frag pb;                                // P^T B-fragment
        {
            const bf16_t* pr = pw + lrow * 32 + half * 8;
            pb.q[0] = *(const u32x4*)(pr);
            pb.q[1] = *(const u32x4*)(pr + 16);
        }

        // ---- prefetch next key-block K tile (no VGPR / counter cost) -------
        // clamp column so the last iteration prefetches a safe in-bounds addr
        const int jpf = (j0 + 32 < NN) ? (j0 + 32) : 0;
        {
            const bf16_t* pk = kbase0 + (size_t)jpf * DK;
            __builtin_prefetch(pk, 0, 3);             // rows jpf+0..15, lo cols
            __builtin_prefetch(pk + 64, 0, 3);        //                 hi cols
            __builtin_prefetch(pk + 16 * DK, 0, 3);   // rows jpf+16..31
            __builtin_prefetch(pk + 16 * DK + 64, 0, 3);
        }

        // ---- rescale accumulators (skip when all alpha == 1.0 exactly) -----
        if (__any(alane != 1.0f)) {
#pragma unroll
            for (int ft = 0; ft < 16; ++ft) {
#pragma unroll
                for (int e = 0; e < 8; ++e) acc[ft][e] *= alane;
            }
        }

        // ---- O^T += V^T P^T over 16 feature tiles (depth-3 pipeline) -------
        // plus one prefetch per tile for the next key-block's V columns
        B16Frag va[3];
        VLOAD(0);
        VLOAD(1);
#pragma unroll
        for (int ft = 0; ft < 16; ++ft) {
            if (ft < 14) VLOAD(ft + 2);
            __builtin_prefetch(vbase + (size_t)(ft * 16) * NN + jpf, 0, 3);
            acc[ft] = __builtin_amdgcn_wmma_f32_16x16x32_bf16(
                false, va[ft % 3].v, false, pb.v, (short)0, acc[ft],
                false, false);
        }
    }

    // ---- final l: cross-lane reduce the per-lane partials, broadcast -------
#pragma unroll
    for (int r = 0; r < 8; ++r) {
        float ls = l_r[r];
        ls += __shfl_xor(ls, 1, 32);
        ls += __shfl_xor(ls, 2, 32);
        ls += __shfl_xor(ls, 4, 32);
        ls += __shfl_xor(ls, 8, 32);
        if (lrow == r) sb[r + 8 * half] = ls;
    }
    const float li = 1.0f / sb[lrow];

    // ---- O^T / l, add residual H, store (contiguous 8 floats per lane) -----
    const float* hres = H   + (size_t)(q0 + lrow) * DV;
    float*       orow = out + (size_t)(q0 + lrow) * DV;
#pragma unroll
    for (int ft = 0; ft < 16; ++ft) {
        const int fb = ft * 16 + 8 * half;   // this lane's 8 features
        const f32x4 h0 = *(const f32x4*)(hres + fb);
        const f32x4 h1 = *(const f32x4*)(hres + fb + 4);
        f32x4 o0, o1;
#pragma unroll
        for (int e = 0; e < 4; ++e) {
            o0[e] = acc[ft][e]     * li + h0[e];
            o1[e] = acc[ft][e + 4] * li + h1[e];
        }
        *(f32x4*)(orow + fb)     = o0;
        *(f32x4*)(orow + fb + 4) = o1;
    }
}

// ---------------------------------------------------------------------------
extern "C" void kernel_launch(void* const* d_in, const int* in_sizes, int n_in,
                              void* d_out, int out_size, void* d_ws,
                              size_t ws_size, hipStream_t stream)
{
    const float* H  = (const float*)d_in[0];
    const float* Wq = (const float*)d_in[1];
    const float* bq = (const float*)d_in[2];
    const float* Wk = (const float*)d_in[3];
    const float* bk = (const float*)d_in[4];
    float* out = (float*)d_out;

    char* ws = (char*)d_ws;
    bf16_t* Qb = (bf16_t*)(ws);                        // 4 MB
    bf16_t* Kb = (bf16_t*)(ws + (4u << 20));           // 4 MB
    bf16_t* HT = (bf16_t*)(ws + (8u << 20));           // 8 MB

    proj_kernel<<<NN, 128, 0, stream>>>(H, Wq, bq, Wk, bk, Qb, Kb);
    transpose_kernel<<<dim3(NN / 64, DV / 64), 256, 0, stream>>>(H, HT);
    attn_kernel<<<NN / 64, 128, 0, stream>>>(H, Qb, Kb, HT, out);
}
